// Model_9251359555877
// MI455X (gfx1250) — compile-verified, matrix operand order checked by reading, not run on hardware
//
#include <hip/hip_runtime.h>
#include <hip/hip_bf16.h>
#include <stdint.h>

typedef __attribute__((ext_vector_type(16))) _Float16 v16h;
typedef __attribute__((ext_vector_type(8)))  float    v8f;
typedef __attribute__((ext_vector_type(2)))  _Float16 h2;

#define BATCH 4096
#define TLEN  200
#define HIDN  80
#define EMBD  40
#define TPAD  208
#define NTILE 13
#define NWAVE 8

// Compiler fence: blocks LICM from hoisting the loop-invariant B-fragment
// loads out of the tile loop (which previously forced ~400 live VGPRs and
// spilled WMMA operands to scratch).
#define LOOP_FENCE() asm volatile("" ::: "memory")

__device__ __forceinline__ float sigmoidf(float x) {
  return 1.0f / (1.0f + __expf(-x));
}

// ---------------- weight pre-swizzle into WMMA B-fragment layout ----------------
// B-matrix (KxN) 16-bit layout per ISA: lane<16 -> N=lane, K-half 0; lane>=16 -> N=lane-16, K-half +8.
// dword d: K = (d<4?0:16) + khalf + 2*(d%4), packed {B[k][n], B[k+1][n]}.

__global__ void swizzle_w1(const float* __restrict__ aw1, uint32_t* __restrict__ bw1) {
  int idx = blockIdx.x * 256 + threadIdx.x;
  if (idx >= 10 * 5 * 32 * 8) return;          // 10 K-chunks x 5 N-tiles x 32 lanes x 8 dwords
  int d    = idx & 7;
  int lane = (idx >> 3) & 31;
  int f    = idx >> 8;                          // fragment id = kc*5 + nc
  int kc = f / 5, nc = f % 5;
  int n     = nc * 16 + (lane & 15);
  int khalf = (lane >> 4) * 8;
  int kk    = ((d & 4) ? 16 : 0) + khalf + 2 * (d & 3);
  int k     = kc * 32 + kk;
  union { _Float16 h[2]; uint32_t u; } pk;
  pk.h[0] = (_Float16)aw1[k * 80 + n];
  pk.h[1] = (_Float16)aw1[(k + 1) * 80 + n];
  bw1[idx] = pk.u;
}

__global__ void swizzle_w2(const float* __restrict__ aw2, uint32_t* __restrict__ bw2) {
  int idx = blockIdx.x * 256 + threadIdx.x;
  if (idx >= 3 * 3 * 32 * 8) return;            // K padded 80->96, N padded 40->48
  int d    = idx & 7;
  int lane = (idx >> 3) & 31;
  int f    = idx >> 8;
  int kc = f / 3, nc = f % 3;
  int n     = nc * 16 + (lane & 15);
  int khalf = (lane >> 4) * 8;
  int kk    = ((d & 4) ? 16 : 0) + khalf + 2 * (d & 3);
  int k     = kc * 32 + kk;
  union { _Float16 h[2]; uint32_t u; } pk;
  pk.h[0] = (k     < 80 && n < 40) ? (_Float16)aw2[k * 40 + n]       : (_Float16)0.0f;
  pk.h[1] = (k + 1 < 80 && n < 40) ? (_Float16)aw2[(k + 1) * 40 + n] : (_Float16)0.0f;
  bw2[idx] = pk.u;
}

// ---------------- main fused DIN forward: one workgroup per batch row ----------------
__global__ __launch_bounds__(256) void din_forward(
    const int* __restrict__ user_his, const int* __restrict__ user_cate,
    const int* __restrict__ target_item, const int* __restrict__ target_cate,
    const int* __restrict__ sl, const float* __restrict__ label,
    const float* __restrict__ item_emb, const float* __restrict__ cate_emb,
    const float* __restrict__ item_b,
    const uint32_t* __restrict__ bw1, const float* __restrict__ ab1,
    const uint32_t* __restrict__ bw2, const float* __restrict__ ab2,
    const float* __restrict__ aw3, const float* __restrict__ ab3,
    const float* __restrict__ ow,  const float* __restrict__ ob,
    const float* __restrict__ dw1, const float* __restrict__ db1,
    const float* __restrict__ dw2, const float* __restrict__ db2,
    const float* __restrict__ dw3, const float* __restrict__ db3,
    float* __restrict__ out_arr, float* __restrict__ block_bce)
{
  __shared__ _Float16 his[TPAD * HIDN];        // 33,280 B  (keys, f16)
  __shared__ _Float16 q16[HIDN];
  __shared__ float    q32[HIDN];
  __shared__ float    aw3s[EMBD];
  __shared__ float    wlds[TPAD];              // attention weights per t
  __shared__ _Float16 s1[NWAVE][16 * 96];      // 24,576 B  (layer-1 act staging, K padded to 96)
  __shared__ float    att[HIDN];
  __shared__ float    ec[2 * HIDN];
  __shared__ float    fc1[HIDN];
  __shared__ float    fc2[EMBD];

  const int b    = blockIdx.x;
  const int tid  = threadIdx.x;
  const int lane = tid & 31;
  const int wave = tid >> 5;

  // ---- query embedding + small vectors ----
  if (tid < EMBD) {
    float v = item_emb[(size_t)target_item[b] * EMBD + tid];
    q32[tid] = v; q16[tid] = (_Float16)v;
  } else if (tid < HIDN) {
    float v = cate_emb[(size_t)target_cate[b] * EMBD + (tid - EMBD)];
    q32[tid] = v; q16[tid] = (_Float16)v;
  } else if (tid < HIDN + EMBD) {
    aw3s[tid - HIDN] = aw3[tid - HIDN];
  }

  // ---- gather history (item||cate) -> f16 LDS, one (row, half) per job ----
  for (int job = tid; job < 2 * TLEN; job += 256) {
    int r = job >> 1;
    const float* src = (job & 1)
        ? (cate_emb + (size_t)user_cate[b * TLEN + r] * EMBD)
        : (item_emb + (size_t)user_his [b * TLEN + r] * EMBD);
    _Float16* dst = &his[r * HIDN + (job & 1) * EMBD];
#pragma unroll
    for (int i = 0; i < 10; i++) {
      float4 v = ((const float4*)src)[i];
      dst[i * 4 + 0] = (_Float16)v.x; dst[i * 4 + 1] = (_Float16)v.y;
      dst[i * 4 + 2] = (_Float16)v.z; dst[i * 4 + 3] = (_Float16)v.w;
    }
  }
  for (int e = tid; e < (TPAD - TLEN) * HIDN; e += 256) his[TLEN * HIDN + e] = (_Float16)0.0f;
  for (int e = tid; e < NWAVE * 16 * 96; e += 256) (&s1[0][0])[e] = (_Float16)0.0f;

  __syncthreads();

  const int slb   = sl[b];
  const int col   = lane & 15;
  const int khalf = (lane >> 4) * 8;

  // ---- per-wave 16-row tiles over T ----
  for (int rt = wave; rt < NTILE; rt += NWAVE) {
    LOOP_FENCE();                              // keep B-fragment loads inside this loop
    const int m = rt * 16 + col;               // A-matrix row for this lane

    // ===== layer 1: [16,320] x [320,80], f16 WMMA, f32 acc, bias-initialized =====
    v8f acc[5];
#pragma unroll
    for (int nc = 0; nc < 5; nc++) {
      float bv = ab1[nc * 16 + col];
#pragma unroll
      for (int j = 0; j < 8; j++) acc[nc][j] = bv;
    }

#pragma unroll
    for (int kc = 0; kc < 10; kc++) {
      // A-fragment: synthesize concat [q | his | q-his | q*his] on the fly from LDS
      union { v16h v; h2 p[8]; } au;
#pragma unroll
      for (int d = 0; d < 8; d++) {
        int kk = ((d & 4) ? 16 : 0) + khalf + 2 * (d & 3);
        int k  = kc * 32 + kk;                 // 0..318, even
        int s  = k / 80;                       // concat segment
        int j  = k - s * 80;                   // even, pair stays in-segment
        h2 qv = *(const h2*)&q16[j];
        h2 hv = *(const h2*)&his[m * HIDN + j];
        h2 r;
        if      (s == 0) r = qv;
        else if (s == 1) r = hv;
        else if (s == 2) r = qv - hv;
        else             r = qv * hv;
        au.p[d] = r;
      }
#pragma unroll
      for (int nc = 0; nc < 5; nc++) {
        union { v16h v; uint4 u[2]; } bu;
        const uint4* bp = (const uint4*)(bw1 + (size_t)((kc * 5 + nc) * 32 + lane) * 8);
        bu.u[0] = bp[0]; bu.u[1] = bp[1];
        acc[nc] = __builtin_amdgcn_wmma_f32_16x16x32_f16(
            false, au.v, false, bu.v, (short)0, acc[nc], false, false);
      }
    }

    // sigmoid -> f16 staging tile (rows x 96, pad cols stay zero)
    _Float16* s1w = &s1[wave][0];
#pragma unroll
    for (int nc = 0; nc < 5; nc++) {
#pragma unroll
      for (int j = 0; j < 8; j++) {
        int row = j + 8 * (lane >> 4);
        s1w[row * 96 + nc * 16 + col] = (_Float16)sigmoidf(acc[nc][j]);
      }
    }

    // ===== layer 2: [16,80] x [80,40] (padded 96x48), f16 WMMA =====
    v8f acc2[3];
#pragma unroll
    for (int nc = 0; nc < 3; nc++) {
      int c = nc * 16 + col;
      float bv = (c < EMBD) ? ab2[c] : 0.0f;
#pragma unroll
      for (int j = 0; j < 8; j++) acc2[nc][j] = bv;
    }
#pragma unroll
    for (int kc = 0; kc < 3; kc++) {
      union { v16h v; h2 p[8]; } au;
#pragma unroll
      for (int d = 0; d < 8; d++) {
        int kk = ((d & 4) ? 16 : 0) + khalf + 2 * (d & 3);
        int k  = kc * 32 + kk;                 // < 96
        au.p[d] = *(const h2*)&s1w[col * 96 + k];
      }
#pragma unroll
      for (int nc = 0; nc < 3; nc++) {
        union { v16h v; uint4 u[2]; } bu;
        const uint4* bp = (const uint4*)(bw2 + (size_t)((kc * 3 + nc) * 32 + lane) * 8);
        bu.u[0] = bp[0]; bu.u[1] = bp[1];
        acc2[nc] = __builtin_amdgcn_wmma_f32_16x16x32_f16(
            false, au.v, false, bu.v, (short)0, acc2[nc], false, false);
      }
    }

    // ===== layer 3: dot over 40 cols via shfl_xor tree across the 16 col-lanes =====
    float part[8];
#pragma unroll
    for (int j = 0; j < 8; j++) part[j] = 0.0f;
#pragma unroll
    for (int nc = 0; nc < 3; nc++) {
      int c = nc * 16 + col;
      float a3 = (c < EMBD) ? aw3s[c] : 0.0f;
#pragma unroll
      for (int j = 0; j < 8; j++) part[j] += sigmoidf(acc2[nc][j]) * a3;
    }
#pragma unroll
    for (int off = 1; off < 16; off <<= 1) {
#pragma unroll
      for (int j = 0; j < 8; j++) part[j] += __shfl_xor(part[j], off, 32);
    }
    if (col == 0) {                            // lanes 0 (rows 0..7) and 16 (rows 8..15)
      float b3 = ab3[0];
#pragma unroll
      for (int j = 0; j < 8; j++) {
        int t = rt * 16 + j + 8 * (lane >> 4);
        float w = sigmoidf(part[j] + b3);
        wlds[t] = (t < slb) ? w : 0.0f;
      }
    }
  }

  __syncthreads();

  // ---- att[h] = sum_t w[t] * his[t][h] ----
  if (tid < HIDN) {
    float s = 0.0f;
    for (int t = 0; t < TLEN; t++) s += wlds[t] * (float)his[t * HIDN + tid];
    att[tid] = s;
  }
  __syncthreads();
  // ---- out_fc = att @ ow + ob ; ec = [out_fc, q] ----
  if (tid < HIDN) {
    float s = ob[tid];
    for (int h = 0; h < HIDN; h++) s += att[h] * ow[h * HIDN + tid];
    ec[tid] = s;
    ec[HIDN + tid] = q32[tid];
  }
  __syncthreads();
  if (tid < HIDN) {
    float s = db1[tid];
    for (int k = 0; k < 2 * HIDN; k++) s += ec[k] * dw1[k * HIDN + tid];
    fc1[tid] = sigmoidf(s);
  }
  __syncthreads();
  if (tid < EMBD) {
    float s = db2[tid];
    for (int k = 0; k < HIDN; k++) s += fc1[k] * dw2[k * EMBD + tid];
    fc2[tid] = sigmoidf(s);
  }
  __syncthreads();
  if (tid == 0) {
    float s = db3[0];
    for (int k = 0; k < EMBD; k++) s += fc2[k] * dw3[k];
    float logit = s + item_b[target_item[b]];
    float p = sigmoidf(logit);
    out_arr[1 + b] = p;
    float pc  = fminf(fmaxf(p, 1e-7f), 1.0f - 1e-7f);
    float lab = label[b];
    block_bce[b] = -(lab * logf(pc) + (1.0f - lab) * log1pf(-pc));
  }
}

// ---------------- deterministic mean of per-block BCE ----------------
__global__ void reduce_bce(const float* __restrict__ bb, float* __restrict__ out_arr) {
  __shared__ float sm[256];
  float s = 0.0f;
  for (int i = threadIdx.x; i < BATCH; i += 256) s += bb[i];
  sm[threadIdx.x] = s;
  __syncthreads();
  for (int off = 128; off > 0; off >>= 1) {
    if (threadIdx.x < off) sm[threadIdx.x] += sm[threadIdx.x + off];
    __syncthreads();
  }
  if (threadIdx.x == 0) out_arr[0] = sm[0] / (float)BATCH;
}

extern "C" void kernel_launch(void* const* d_in, const int* in_sizes, int n_in,
                              void* d_out, int out_size, void* d_ws, size_t ws_size,
                              hipStream_t stream) {
  const int*   user_his    = (const int*)  d_in[0];
  const int*   user_cate   = (const int*)  d_in[1];
  const int*   target_item = (const int*)  d_in[2];
  const int*   target_cate = (const int*)  d_in[3];
  const int*   sl          = (const int*)  d_in[4];
  const float* label       = (const float*)d_in[5];
  const float* item_emb    = (const float*)d_in[6];
  const float* cate_emb    = (const float*)d_in[7];
  const float* item_b      = (const float*)d_in[8];
  const float* aw1 = (const float*)d_in[9];   const float* ab1 = (const float*)d_in[10];
  const float* aw2 = (const float*)d_in[11];  const float* ab2 = (const float*)d_in[12];
  const float* aw3 = (const float*)d_in[13];  const float* ab3 = (const float*)d_in[14];
  const float* ow  = (const float*)d_in[15];  const float* ob  = (const float*)d_in[16];
  const float* dw1 = (const float*)d_in[17];  const float* db1 = (const float*)d_in[18];
  const float* dw2 = (const float*)d_in[19];  const float* db2 = (const float*)d_in[20];
  const float* dw3 = (const float*)d_in[21];  const float* db3 = (const float*)d_in[22];

  // workspace layout: bw1 (12800 dwords) | bw2 (2304 dwords) | block_bce (4096 f32)
  uint32_t* bw1 = (uint32_t*)d_ws;
  uint32_t* bw2 = bw1 + 12800;
  float* block_bce = (float*)(bw2 + 2304);
  float* out_arr = (float*)d_out;             // [0]=ave_loss, [1..4096]=p

  swizzle_w1<<<50, 256, 0, stream>>>(aw1, bw1);
  swizzle_w2<<<9,  256, 0, stream>>>(aw2, bw2);
  din_forward<<<BATCH, 256, 0, stream>>>(
      user_his, user_cate, target_item, target_cate, sl, label,
      item_emb, cate_emb, item_b,
      bw1, ab1, bw2, ab2, aw3, ab3,
      ow, ob, dw1, db1, dw2, db2, dw3, db3,
      out_arr, block_bce);
  reduce_bce<<<1, 256, 0, stream>>>(block_bce, out_arr);
}
